// INTagger_39058432589886
// MI455X (gfx1250) — matrix operand order repaired and checked by reading, not compiled, not run on hardware
//
#include <hip/hip_runtime.h>

typedef __attribute__((ext_vector_type(16))) _Float16 v16h;
typedef __attribute__((ext_vector_type(8)))  _Float16 v8h;
typedef __attribute__((ext_vector_type(8)))  float    v8f;

namespace {
constexpr int NPART = 60;   // N
constexpr int NVERT = 5;    // NV
constexpr int PFEAT = 30;   // P
constexpr int SFEAT = 14;   // S
constexpr int HID   = 60;
constexpr int DE    = 20;
constexpr int DO    = 24;
constexpr int NCLS  = 6;
constexpr int NPAIR = NPART * (NPART - 1);  // 3540
constexpr int NTRK  = NPART * NVERT;        // 300
constexpr int BLK   = 256;
constexpr int NWAVE = BLK / 32;

// LDS row strides (halves), chosen non-power-of-2 to spread the 64 x 4B banks:
// 40 halves = 80B  -> bank offset 20*r mod 64: 16 distinct groups for r=0..15
// 72 halves = 144B -> bank offset 36*r mod 64: 16 distinct groups for r=0..15
constexpr int XS = 40;   // Xh / Yh / Ebar-f16 row stride (K=32 panels)
constexpr int HS = 72;   // hidden panel row stride (K=64 panels)
constexpr int W1S = 40;  // weight din_pad for K=32 weights
constexpr int W2S = 72;  // weight din_pad for K=64 weights
}

// ---- WMMA fragment helpers (layouts per CDNA5 ISA 7.12.2, wave32) ----

// A-matrix 16x32 f16 fragment. rowk points at this row's K-block base (16B-aligned).
// lane 0-15: halves 0-7 = K 0-7, halves 8-15 = K 16-23
// lane 16-31: halves 0-7 = K 8-15, halves 8-15 = K 24-31
__device__ __forceinline__ v16h ldA(const _Float16* rowk, int lane) {
  const int g = (lane >> 4) & 1;
  v8h lo = *(const v8h*)(rowk + 8 * g);
  v8h hi = *(const v8h*)(rowk + 16 + 8 * g);
  v16h r;
#pragma unroll
  for (int i = 0; i < 8; ++i) { r[i] = lo[i]; r[i + 8] = hi[i]; }
  return r;
}

// B-matrix 32x16 f16 fragment from N-major weight tile W[dout_pad][dpi]:
// lane = N (n0 + lane&15), halves 0-7 = K k0+8*(lane>>4)+0..7, halves 8-15 = K k0+16+8*(lane>>4)+0..7. Two 16B loads.
__device__ __forceinline__ v16h ldB(const _Float16* W, int dpi, int n0, int k0, int lane) {
  const _Float16* p = W + (size_t)(n0 + (lane & 15)) * dpi + k0 + 8 * ((lane >> 4) & 1);
  v8h lo = *(const v8h*)p;
  v8h hi = *(const v8h*)(p + 16);
  v16h r;
#pragma unroll
  for (int i = 0; i < 8; ++i) { r[i] = lo[i]; r[i + 8] = hi[i]; }
  return r;
}

__device__ __forceinline__ v8f wmma_f16(v16h a, v16h b, v8f c) {
  return __builtin_amdgcn_wmma_f32_16x16x32_f16(false, a, false, b, (short)0, c, false, false);
}

// Store f32 D tile (bias + relu + f16 convert) into a [rows][HS] LDS panel.
// D layout: element i of lane -> row = i + 8*(lane>>4), col = lane&15 (+n0).
__device__ __forceinline__ void storeH(_Float16* H, v8f acc, int n0, const float* bias,
                                       int nvalid, int lane) {
  const int col = n0 + (lane & 15);
  const int rb  = 8 * ((lane >> 4) & 1);
#pragma unroll
  for (int i = 0; i < 8; ++i) {
    float v = (col < nvalid) ? acc[i] + bias[col] : 0.f;
    v = v > 0.f ? v : 0.f;
    H[(i + rb) * HS + col] = (_Float16)v;
  }
}

// Stage weight slice src[dout, din_tot][:, c0:c0+ncol] into f16 LDS tile [dpo][dpi], zero-padded.
__device__ __forceinline__ void stage_w(_Float16* dst, const float* src, int dout, int din_tot,
                                        int c0, int ncol, int dpo, int dpi) {
  for (int i = threadIdx.x; i < dpo * dpi; i += BLK) {
    int r = i / dpi, c = i % dpi;
    dst[i] = (_Float16)((r < dout && c < ncol) ? src[r * din_tot + c0 + c] : 0.f);
  }
}
__device__ __forceinline__ void stage_b(float* dst, const float* src, int n, int npad) {
  for (int i = threadIdx.x; i < npad; i += BLK) dst[i] = (i < n) ? src[i] : 0.f;
}

__global__ __launch_bounds__(BLK) void in_tagger_fused(
    const float* __restrict__ x, const float* __restrict__ y,
    const float* fr1_w, const float* fr1_b, const float* fr2_w, const float* fr2_b,
    const float* fr3_w, const float* fr3_b,
    const float* fp1_w, const float* fp1_b, const float* fp2_w, const float* fp2_b,
    const float* fp3_w, const float* fp3_b,
    const float* fo1_w, const float* fo1_b, const float* fo2_w, const float* fo2_b,
    const float* fo3_w, const float* fo3_b,
    const float* fc_w, const float* fc_b,
    float* __restrict__ out)
{
  __shared__ _Float16 Xh[64 * XS];           // xt[b] rows, padded
  __shared__ _Float16 Yh[16 * XS];           // yt[b] rows, padded
  __shared__ _Float16 Warena[14592];         // weight arena, re-staged per phase
  __shared__ float    Barena[192];           // bias arena (64+64+32)
  __shared__ float    ebarPP[64 * DE];
  __shared__ float    ebarPV[64 * DE];
  __shared__ _Float16 EbarPPh[64 * XS];
  __shared__ _Float16 EbarPVh[64 * XS];
  __shared__ _Float16 H1s[NWAVE][32 * HS];   // per-wave hidden panels (M=32)
  __shared__ _Float16 H2s[NWAVE][32 * HS];
  __shared__ float    Osum[DO];

  const int b    = blockIdx.x;
  const int tid  = threadIdx.x;
  const int lane = tid & 31;
  const int wv   = tid >> 5;
  _Float16* h1 = H1s[wv];
  _Float16* h2 = H2s[wv];
  float* Bb1 = Barena;
  float* Bb2 = Barena + 64;
  float* Bb3 = Barena + 128;

  // ---------- phase 0: stage inputs + pair-MLP weights ----------
  for (int i = tid; i < 64 * XS; i += BLK) {
    int n = i / XS, c = i % XS;
    Xh[i] = (_Float16)((n < NPART && c < PFEAT) ? x[((size_t)b * PFEAT + c) * NPART + n] : 0.f);
  }
  for (int i = tid; i < 16 * XS; i += BLK) {
    int v = i / XS, c = i % XS;
    Yh[i] = (_Float16)((v < NVERT && c < SFEAT) ? y[((size_t)b * SFEAT + c) * NVERT + v] : 0.f);
  }
  for (int i = tid; i < 64 * DE; i += BLK) { ebarPP[i] = 0.f; ebarPV[i] = 0.f; }
  if (tid < DO) Osum[tid] = 0.f;

  _Float16* Wa  = Warena;                 // 64 x W1S
  _Float16* Wb  = Warena + 64 * W1S;      // 64 x W1S
  _Float16* Wl2 = Warena + 2 * 64 * W1S;  // 64 x W2S
  _Float16* Wl3 = Wl2 + 64 * W2S;         // 32 x W2S
  stage_w(Wa,  fr1_w, HID, 2 * PFEAT, 0,     PFEAT, 64, W1S);
  stage_w(Wb,  fr1_w, HID, 2 * PFEAT, PFEAT, PFEAT, 64, W1S);
  stage_w(Wl2, fr2_w, HID, HID, 0, HID, 64, W2S);
  stage_w(Wl3, fr3_w, DE,  HID, 0, HID, 32, W2S);
  stage_b(Bb1, fr1_b, HID, 64);
  stage_b(Bb2, fr2_b, HID, 64);
  stage_b(Bb3, fr3_b, DE, 32);
  __syncthreads();

  // ---------- phase 1: pair MLP over 3540 rows, M=32 per wave iteration ----------
  {
    const int ntile2 = (NPAIR + 31) / 32;  // 111
    for (int t = wv; t < ntile2; t += NWAVE) {
      const int base = t * 32;
      int p0 = base + (lane & 15);
      int p1 = p0 + 16;
      int r0 = 63, s0 = 63, r1 = 63, s1 = 63;   // rows 60..63 of Xh are zero
      if (p0 < NPAIR) { r0 = p0 / 59; int m = p0 - r0 * 59; s0 = m + (m >= r0 ? 1 : 0); }
      if (p1 < NPAIR) { r1 = p1 / 59; int m = p1 - r1 * 59; s1 = m + (m >= r1 ? 1 : 0); }
      v16h aR0 = ldA(Xh + r0 * XS, lane);
      v16h aS0 = ldA(Xh + s0 * XS, lane);
      v16h aR1 = ldA(Xh + r1 * XS, lane);
      v16h aS1 = ldA(Xh + s1 * XS, lane);
      // L1: share B fragments across both M-tiles
#pragma unroll
      for (int nt = 0; nt < 4; ++nt) {
        v16h b0 = ldB(Wa, W1S, nt * 16, 0, lane);
        v16h b1 = ldB(Wb, W1S, nt * 16, 0, lane);
        v8f acc0 = {}, acc1 = {};
        acc0 = wmma_f16(aR0, b0, acc0);
        acc1 = wmma_f16(aR1, b0, acc1);
        acc0 = wmma_f16(aS0, b1, acc0);
        acc1 = wmma_f16(aS1, b1, acc1);
        storeH(h1,           acc0, nt * 16, Bb1, HID, lane);
        storeH(h1 + 16 * HS, acc1, nt * 16, Bb1, HID, lane);
      }
      // L2: hoist A fragments, reuse across 4 N-tiles
      {
        const int row = (lane & 15);
        v16h a00 = ldA(h1 + row * HS, lane);
        v16h a01 = ldA(h1 + row * HS + 32, lane);
        v16h a10 = ldA(h1 + (16 + row) * HS, lane);
        v16h a11 = ldA(h1 + (16 + row) * HS + 32, lane);
#pragma unroll
        for (int nt = 0; nt < 4; ++nt) {
          v16h b0 = ldB(Wl2, W2S, nt * 16, 0, lane);
          v16h b1 = ldB(Wl2, W2S, nt * 16, 32, lane);
          v8f acc0 = {}, acc1 = {};
          acc0 = wmma_f16(a00, b0, acc0);
          acc1 = wmma_f16(a10, b0, acc1);
          acc0 = wmma_f16(a01, b1, acc0);
          acc1 = wmma_f16(a11, b1, acc1);
          storeH(h2,           acc0, nt * 16, Bb2, HID, lane);
          storeH(h2 + 16 * HS, acc1, nt * 16, Bb2, HID, lane);
        }
      }
      // L3 + scatter-add into ebarPP[RECV]
      {
        const int row = (lane & 15);
        v16h a00 = ldA(h2 + row * HS, lane);
        v16h a01 = ldA(h2 + row * HS + 32, lane);
        v16h a10 = ldA(h2 + (16 + row) * HS, lane);
        v16h a11 = ldA(h2 + (16 + row) * HS + 32, lane);
#pragma unroll
        for (int nt = 0; nt < 2; ++nt) {
          v16h b0 = ldB(Wl3, W2S, nt * 16, 0, lane);
          v16h b1 = ldB(Wl3, W2S, nt * 16, 32, lane);
          v8f acc0 = {}, acc1 = {};
          acc0 = wmma_f16(a00, b0, acc0);
          acc1 = wmma_f16(a10, b0, acc1);
          acc0 = wmma_f16(a01, b1, acc0);
          acc1 = wmma_f16(a11, b1, acc1);
          int col = nt * 16 + (lane & 15);
          if (col < DE) {
            const int rb = 8 * ((lane >> 4) & 1);
#pragma unroll
            for (int i = 0; i < 8; ++i) {
              int pp = base + i + rb;
              if (pp < NPAIR) {
                float v = acc0[i] + Bb3[col];
                atomicAdd(&ebarPP[(pp / 59) * DE + col], v > 0.f ? v : 0.f);
              }
              int pq = base + 16 + i + rb;
              if (pq < NPAIR) {
                float v = acc1[i] + Bb3[col];
                atomicAdd(&ebarPP[(pq / 59) * DE + col], v > 0.f ? v : 0.f);
              }
            }
          }
        }
      }
    }
  }
  __syncthreads();

  // ---------- phase 2: stage pv weights, then pv MLP over 300 rows (M=16) ----------
  stage_w(Wa,  fp1_w, HID, PFEAT + SFEAT, 0,     PFEAT, 64, W1S);
  stage_w(Wb,  fp1_w, HID, PFEAT + SFEAT, PFEAT, SFEAT, 64, W1S);
  stage_w(Wl2, fp2_w, HID, HID, 0, HID, 64, W2S);
  stage_w(Wl3, fp3_w, DE,  HID, 0, HID, 32, W2S);
  stage_b(Bb1, fp1_b, HID, 64);
  stage_b(Bb2, fp2_b, HID, 64);
  stage_b(Bb3, fp3_b, DE, 32);
  __syncthreads();
  {
    const int ntile = (NTRK + 15) / 16;  // 19
    for (int t = wv; t < ntile; t += NWAVE) {
      int p = t * 16 + (lane & 15);
      int k = 63, vv = 15;               // zero rows
      if (p < NTRK) { k = p / NVERT; vv = p - k * NVERT; }
      v16h aX = ldA(Xh + k * XS, lane);
      v16h aY = ldA(Yh + vv * XS, lane);
#pragma unroll
      for (int nt = 0; nt < 4; ++nt) {
        v8f acc = {};
        acc = wmma_f16(aX, ldB(Wa, W1S, nt * 16, 0, lane), acc);
        acc = wmma_f16(aY, ldB(Wb, W1S, nt * 16, 0, lane), acc);
        storeH(h1, acc, nt * 16, Bb1, HID, lane);
      }
      {
        const int row = (lane & 15);
        v16h a0 = ldA(h1 + row * HS, lane);
        v16h a1 = ldA(h1 + row * HS + 32, lane);
#pragma unroll
        for (int nt = 0; nt < 4; ++nt) {
          v8f acc = {};
          acc = wmma_f16(a0, ldB(Wl2, W2S, nt * 16, 0, lane), acc);
          acc = wmma_f16(a1, ldB(Wl2, W2S, nt * 16, 32, lane), acc);
          storeH(h2, acc, nt * 16, Bb2, HID, lane);
        }
      }
      {
        const int row = (lane & 15);
        v16h a0 = ldA(h2 + row * HS, lane);
        v16h a1 = ldA(h2 + row * HS + 32, lane);
#pragma unroll
        for (int nt = 0; nt < 2; ++nt) {
          v8f acc = {};
          acc = wmma_f16(a0, ldB(Wl3, W2S, nt * 16, 0, lane), acc);
          acc = wmma_f16(a1, ldB(Wl3, W2S, nt * 16, 32, lane), acc);
          int col = nt * 16 + (lane & 15);
          if (col < DE) {
            const int rb = 8 * ((lane >> 4) & 1);
#pragma unroll
            for (int i = 0; i < 8; ++i) {
              int pp = t * 16 + i + rb;
              if (pp < NTRK) {
                float v = acc[i] + Bb3[col];
                atomicAdd(&ebarPV[(pp / NVERT) * DE + col], v > 0.f ? v : 0.f);
              }
            }
          }
        }
      }
    }
  }
  __syncthreads();

  // ---------- phase 3: convert ebar to f16 panels + stage object weights ----------
  for (int i = tid; i < 64 * XS; i += BLK) {
    int n = i / XS, c = i % XS;
    bool ok = (n < NPART && c < DE);
    EbarPPh[i] = (_Float16)(ok ? ebarPP[n * DE + c] : 0.f);
    EbarPVh[i] = (_Float16)(ok ? ebarPV[n * DE + c] : 0.f);
  }
  _Float16* Wo1a = Warena;                 // 64 x W1S
  _Float16* Wo1b = Warena + 64 * W1S;      // 64 x W1S
  _Float16* Wo1c = Warena + 2 * 64 * W1S;  // 64 x W1S
  _Float16* Wo2  = Warena + 3 * 64 * W1S;  // 64 x W2S
  _Float16* Wo3  = Wo2 + 64 * W2S;         // 32 x W2S
  stage_w(Wo1a, fo1_w, HID, PFEAT + 2 * DE, 0,          PFEAT, 64, W1S);
  stage_w(Wo1b, fo1_w, HID, PFEAT + 2 * DE, PFEAT,      DE,    64, W1S);
  stage_w(Wo1c, fo1_w, HID, PFEAT + 2 * DE, PFEAT + DE, DE,    64, W1S);
  stage_w(Wo2,  fo2_w, HID, HID, 0, HID, 64, W2S);
  stage_w(Wo3,  fo3_w, DO,  HID, 0, HID, 32, W2S);
  stage_b(Bb1, fo1_b, HID, 64);
  stage_b(Bb2, fo2_b, HID, 64);
  stage_b(Bb3, fo3_b, DO, 32);
  __syncthreads();

  // ---------- phase 4: object MLP over 60 particle rows (waves 0..3) ----------
  if (wv < 4) {
    const int t = wv;
    const int n = t * 16 + (lane & 15);  // up to 63; rows 60..63 are zero
    v16h aX = ldA(Xh      + n * XS, lane);
    v16h aP = ldA(EbarPPh + n * XS, lane);
    v16h aV = ldA(EbarPVh + n * XS, lane);
#pragma unroll
    for (int nt = 0; nt < 4; ++nt) {
      v8f acc = {};
      acc = wmma_f16(aX, ldB(Wo1a, W1S, nt * 16, 0, lane), acc);
      acc = wmma_f16(aP, ldB(Wo1b, W1S, nt * 16, 0, lane), acc);
      acc = wmma_f16(aV, ldB(Wo1c, W1S, nt * 16, 0, lane), acc);
      storeH(h1, acc, nt * 16, Bb1, HID, lane);
    }
    {
      const int row = (lane & 15);
      v16h a0 = ldA(h1 + row * HS, lane);
      v16h a1 = ldA(h1 + row * HS + 32, lane);
#pragma unroll
      for (int nt = 0; nt < 4; ++nt) {
        v8f acc = {};
        acc = wmma_f16(a0, ldB(Wo2, W2S, nt * 16, 0, lane), acc);
        acc = wmma_f16(a1, ldB(Wo2, W2S, nt * 16, 32, lane), acc);
        storeH(h2, acc, nt * 16, Bb2, HID, lane);
      }
    }
    {
      const int row = (lane & 15);
      v16h a0 = ldA(h2 + row * HS, lane);
      v16h a1 = ldA(h2 + row * HS + 32, lane);
#pragma unroll
      for (int nt = 0; nt < 2; ++nt) {
        v8f acc = {};
        acc = wmma_f16(a0, ldB(Wo3, W2S, nt * 16, 0, lane), acc);
        acc = wmma_f16(a1, ldB(Wo3, W2S, nt * 16, 32, lane), acc);
        int col = nt * 16 + (lane & 15);
        if (col < DO) {
          const int rb = 8 * ((lane >> 4) & 1);
#pragma unroll
          for (int i = 0; i < 8; ++i) {
            int nn = t * 16 + i + rb;
            if (nn < NPART) {
              float v = acc[i] + Bb3[col];
              atomicAdd(&Osum[col], v > 0.f ? v : 0.f);  // sum-pool over particles
            }
          }
        }
      }
    }
  }
  __syncthreads();

  // ---------- phase 5: final classifier ----------
  if (tid < NCLS) {
    float acc = fc_b[tid];
#pragma unroll
    for (int d = 0; d < DO; ++d) acc += Osum[d] * fc_w[tid * DO + d];
    out[(size_t)b * NCLS + tid] = acc;
  }
}

extern "C" void kernel_launch(void* const* d_in, const int* in_sizes, int n_in,
                              void* d_out, int out_size, void* d_ws, size_t ws_size,
                              hipStream_t stream) {
  (void)n_in; (void)out_size; (void)d_ws; (void)ws_size;
  const float* x = (const float*)d_in[0];
  const float* y = (const float*)d_in[1];
  const int batch = in_sizes[0] / (PFEAT * NPART);
  in_tagger_fused<<<batch, BLK, 0, stream>>>(
      x, y,
      (const float*)d_in[2],  (const float*)d_in[3],
      (const float*)d_in[4],  (const float*)d_in[5],
      (const float*)d_in[6],  (const float*)d_in[7],
      (const float*)d_in[8],  (const float*)d_in[9],
      (const float*)d_in[10], (const float*)d_in[11],
      (const float*)d_in[12], (const float*)d_in[13],
      (const float*)d_in[14], (const float*)d_in[15],
      (const float*)d_in[16], (const float*)d_in[17],
      (const float*)d_in[18], (const float*)d_in[19],
      (const float*)d_in[20], (const float*)d_in[21],
      (float*)d_out);
}
